// OuterProductMean_77017353552610
// MI455X (gfx1250) — matrix operand order repaired and checked by reading, not compiled, not run on hardware
//
#include <hip/hip_runtime.h>

// ---------------------------------------------------------------------------
// OuterProductMean (AlphaFold-style), MI455X / gfx1250, wave32 + WMMA bf16.
//
//   m (S=128, I=256, CM=256) f32 -> LayerNorm -> a,b (S,I,32) bf16
//   o = sum_s outer(a,b)  : GEMM1  M=(i,a)=8192, N=(j,b)=8192, K=s=128
//   z = o . Wout^T        : GEMM2  M=(i,j),     N=z=128,      K=(a,b)=1024
//   o never touches HBM: per-workgroup 32x512 f32 tile -> bf16 LDS tile.
//
// Fragment-major layouts: every WMMA operand = two b128 loads per lane.
//   aT   [(i*32+a)][s]                A-frags: 2x uint4 at half*16 / half*16+8
//   bpack[kb=s/32][(j*32+b)][s%32]    B-frags: 16 contiguous bf16 at half*16
//   wpack[kb=k/32][z][k%32], k=b*32+a same
//   o_lds[jj][b*32+a]                 one ds_store_b128 per 16x16 tile
// Dual accumulator chains per wave hide the WMMA->WMMA hazard NOPs.
// ---------------------------------------------------------------------------

#define NSEQ 128
#define NRES 256
#define CM   256
#define CC   32        // projection dim
#define CZ   128       // output dim z
#define KAB  (CC*CC)   // 1024
#define NB   (NRES*CC) // 8192 (j,b) columns

typedef __attribute__((ext_vector_type(16))) __bf16 v16bf;
typedef __attribute__((ext_vector_type(8)))  float  v8f;

union FragAB {
  v16bf          v;
  unsigned short us[16];
  unsigned       u[8];
  uint4          q[2];
};

__device__ __forceinline__ unsigned short f32_to_bf16_bits(float f) {
  unsigned u = __builtin_bit_cast(unsigned, f);
  unsigned r = u + 0x7FFFu + ((u >> 16) & 1u);   // round-to-nearest-even
  return (unsigned short)(r >> 16);
}

#if __has_builtin(__builtin_amdgcn_cvt_pk_bf16_f32)
__device__ __forceinline__ unsigned pack2_bf16(float lo, float hi) {
  auto r = __builtin_amdgcn_cvt_pk_bf16_f32(lo, hi);   // v_cvt_pk_bf16_f32
  static_assert(sizeof(r) == 4, "cvt_pk_bf16 must be 32-bit");
  return __builtin_bit_cast(unsigned, r);
}
#else
__device__ __forceinline__ unsigned pack2_bf16(float lo, float hi) {
  return (unsigned)f32_to_bf16_bits(lo) |
         ((unsigned)f32_to_bf16_bits(hi) << 16);
}
#endif

__device__ __forceinline__ uint4 pack8_bf16(const v8f& a) {
  uint4 pk;
  pk.x = pack2_bf16(a[0], a[1]);
  pk.y = pack2_bf16(a[2], a[3]);
  pk.z = pack2_bf16(a[4], a[5]);
  pk.w = pack2_bf16(a[6], a[7]);
  return pk;
}

// A-frag (16x32 bf16): lane m=lane&15, half=lane>>4.
// Elements are the contiguous run {half*16 .. +15} of the 32-wide slice.
__device__ __forceinline__ void load_frag_a(FragAB& F, const unsigned short* p) {
  F.q[0] = *(const uint4*)(p);
  F.q[1] = *(const uint4*)(p + 8);
}
// B-frag from fragment-major pack: lane n=lane&15, K(e)=16*half+e -> one
// contiguous 32-byte run.  p = pack_base(n) + half*16
__device__ __forceinline__ void load_frag_bp(FragAB& F, const unsigned short* p) {
  F.q[0] = *(const uint4*)(p);
  F.q[1] = *(const uint4*)(p + 8);
}

__device__ __forceinline__ v8f wmma_bf16(const FragAB& A, const FragAB& B,
                                         v8f C) {
  return __builtin_amdgcn_wmma_f32_16x16x32_bf16(false, A.v, false, B.v,
                                                 (short)0, C, false, false);
}

// ---------------------------------------------------------------------------
// Kernel 0: Wout f32 [z][a*32+b] -> wpack bf16 [k/32][z][k%32], k = b*32+a
// ---------------------------------------------------------------------------
__global__ __launch_bounds__(256) void pack_wout(
    const float* __restrict__ Wout, unsigned short* __restrict__ wpack) {
  int idx = blockIdx.x * 256 + threadIdx.x;   // 0 .. 1024*128-1
  int k = idx >> 7;                           // k = b*32 + a
  int z = idx & (CZ - 1);
  int a = k & 31, b = k >> 5;
  wpack[((size_t)(k >> 5) * CZ + z) * 32 + (k & 31)] =
      f32_to_bf16_bits(Wout[(size_t)z * KAB + a * CC + b]);
}

// ---------------------------------------------------------------------------
// Kernel 1: fused LayerNorm + projections. One wave32 per (s,i) row.
// ---------------------------------------------------------------------------
__global__ __launch_bounds__(256) void opm_ln_proj(
    const float* __restrict__ m, const float* __restrict__ gamma,
    const float* __restrict__ beta, const float* __restrict__ W1,
    const float* __restrict__ b1, const float* __restrict__ W2,
    const float* __restrict__ b2,
    unsigned short* __restrict__ aT,      // [(i*32+a)][128]
    unsigned short* __restrict__ bpack) { // [s/32][(j*32+b)][s%32]
  __shared__ float mn[8][CM];             // per-wave normalized row, 8 KB
  const int wave = threadIdx.x >> 5;
  const int lane = threadIdx.x & 31;
  const int row  = blockIdx.x * 8 + wave; // row = s*NRES + i
  const int s    = row >> 8;
  const int i    = row & (NRES - 1);

  const float* mrow = m + (size_t)row * CM;
  float4 x0 = ((const float4*)mrow)[lane * 2 + 0];
  float4 x1 = ((const float4*)mrow)[lane * 2 + 1];
  float v0[8] = {x0.x, x0.y, x0.z, x0.w, x1.x, x1.y, x1.z, x1.w};

  float sum = 0.f, sq = 0.f;
#pragma unroll
  for (int k = 0; k < 8; ++k) { sum += v0[k]; sq += v0[k] * v0[k]; }
#pragma unroll
  for (int o = 16; o >= 1; o >>= 1) {
    sum += __shfl_xor(sum, o, 32);
    sq  += __shfl_xor(sq,  o, 32);
  }
  const float mu  = sum * (1.0f / CM);
  const float var = fmaxf(sq * (1.0f / CM) - mu * mu, 0.0f);
  const float rs  = rsqrtf(var + 1e-5f);

  float* dst = mn[wave];
  const int c0 = lane * 8;
#pragma unroll
  for (int k = 0; k < 8; ++k) {
    int c = c0 + k;
    dst[c] = (v0[k] - mu) * rs * gamma[c] + beta[c];
  }
  __syncthreads();

  // lane == output channel d (0..31)
  float a_acc = b1[lane];
  float b_acc = b2[lane];
  const float* w1r = W1 + lane * CM;
  const float* w2r = W2 + lane * CM;
#pragma unroll 8
  for (int c = 0; c < CM; ++c) {
    float v = dst[c];
    a_acc = fmaf(v, w1r[c], a_acc);
    b_acc = fmaf(v, w2r[c], b_acc);
  }
  aT[(size_t)(i * CC + lane) * NSEQ + s] = f32_to_bf16_bits(a_acc);
  bpack[((size_t)(s >> 5) * NB + i * CC + lane) * 32 + (s & 31)] =
      f32_to_bf16_bits(b_acc);
}

// ---------------------------------------------------------------------------
// Kernel 2: fused GEMM1 (K=s) + GEMM2 (K=(a,b)) per (i, 16 j's).
// grid = (16 jblocks, 256 i), 256 threads = 8 waves.
// ---------------------------------------------------------------------------
__global__ __launch_bounds__(256) void opm_gemm(
    const unsigned short* __restrict__ aT,     // [(i,a)=8192][128]
    const unsigned short* __restrict__ bpack,  // [4][8192][32]
    const unsigned short* __restrict__ wpack,  // [32][128][32]
    const float* __restrict__ bout,
    float* __restrict__ out) {                 // [i][j][z]
  __shared__ unsigned short o_lds[16 * KAB];   // [jj][b*32+a] bf16, 32 KB

  const int wave   = threadIdx.x >> 5;
  const int lane   = threadIdx.x & 31;
  const int half   = lane >> 4;
  const int l15    = lane & 15;
  const int i      = blockIdx.y;
  const int jblock = blockIdx.x;               // 16 j's per block

  // ---- GEMM1: o[32 x 512] = aT_i (32x128) * bpack slice (128x512) ----
  // Per wave: fixed M-tile (mt), 8 N-tiles done 2-at-a-time (dual acc chains),
  // 4 K-steps. A-frags hoisted once into registers.
  const int mt     = wave >> 2;                // 0..1
  const int ntbase = (wave & 3) * 8;           // 0,8,16,24

  FragAB Af[4];
#pragma unroll
  for (int ks = 0; ks < 4; ++ks)
    load_frag_a(Af[ks], aT + (size_t)(i * CC + mt * 16 + l15) * NSEQ +
                            ks * 32 + half * 16);

#pragma unroll 1
  for (int t = 0; t < 8; t += 2) {
    const int n0 = (ntbase + t) * 16 + l15;
    const int n1 = n0 + 16;
    const unsigned short* bp0 =
        bpack + ((size_t)(jblock * 512 + n0)) * 32 + half * 16;
    const unsigned short* bp1 = bp0 + 16 * 32;
    v8f acc0 = {}, acc1 = {};
#pragma unroll
    for (int ks = 0; ks < 4; ++ks) {
      FragAB B0, B1;
      load_frag_bp(B0, bp0 + (size_t)ks * NB * 32);
      load_frag_bp(B1, bp1 + (size_t)ks * NB * 32);
      acc0 = wmma_bf16(Af[ks], B0, acc0);      // two independent chains ->
      acc1 = wmma_bf16(Af[ks], B1, acc1);      // scheduler interleaves, no NOPs
    }
    // One b128 store per tile: 8 results contiguous in 'a' under k = b*32+a.
    const int base0 = (n0 >> 5) * KAB + (n0 & 31) * CC + mt * 16 + half * 8;
    const int base1 = (n1 >> 5) * KAB + (n1 & 31) * CC + mt * 16 + half * 8;
    *(uint4*)&o_lds[base0] = pack8_bf16(acc0);
    *(uint4*)&o_lds[base1] = pack8_bf16(acc1);
  }
  __syncthreads();

  // ---- GEMM2: z[16 x 128] = o_lds (16x1024) * wpack (1024x128) ----
  // Dual accumulators over even/odd K-steps break the 32-long WMMA chain.
  const int zt = wave;                         // wave owns 16 z's
  const unsigned short* arow = o_lds + l15 * KAB + half * 16;
  const unsigned short* wcol =
      wpack + ((size_t)(zt * 16 + l15)) * 32 + half * 16;
  v8f acc0 = {}, acc1 = {};
#pragma unroll 1
  for (int ks = 0; ks < 32; ks += 2) {
    FragAB A0, B0, A1, B1;
    load_frag_a(A0, arow + ks * 32);
    load_frag_a(A1, arow + ks * 32 + 32);
    load_frag_bp(B0, wcol + (size_t)ks * CZ * 32);
    load_frag_bp(B1, wcol + (size_t)(ks + 1) * CZ * 32);
    acc0 = wmma_bf16(A0, B0, acc0);
    acc1 = wmma_bf16(A1, B1, acc1);
  }
  v8f acc = acc0 + acc1;
#pragma unroll
  for (int r = 0; r < 8; ++r) {
    const int jj = half * 8 + r;
    const int z  = zt * 16 + l15;
    const int j  = jblock * 16 + jj;
    out[((size_t)i * NRES + j) * CZ + z] =
        (acc[r] + bout[z]) * (1.0f / (float)NSEQ);
  }
}

// ---------------------------------------------------------------------------
// Launch. Inputs: 0:m 1:ln_gamma 2:ln_beta 3:W1 4:b1 5:W2 6:b2 7:Wout 8:bout
// Workspace: aT (2 MB) | bpack (2 MB) | wpack (256 KB)  ~= 4.5 MB
// ---------------------------------------------------------------------------
extern "C" void kernel_launch(void* const* d_in, const int* in_sizes, int n_in,
                              void* d_out, int out_size, void* d_ws,
                              size_t ws_size, hipStream_t stream) {
  const float* m     = (const float*)d_in[0];
  const float* gamma = (const float*)d_in[1];
  const float* beta  = (const float*)d_in[2];
  const float* W1    = (const float*)d_in[3];
  const float* b1    = (const float*)d_in[4];
  const float* W2    = (const float*)d_in[5];
  const float* b2    = (const float*)d_in[6];
  const float* Wout  = (const float*)d_in[7];
  const float* bout  = (const float*)d_in[8];
  float* out         = (float*)d_out;

  unsigned short* aT    = (unsigned short*)d_ws;          // 8192*128
  unsigned short* bpack = aT + (size_t)NRES * CC * NSEQ;  // 4*8192*32
  unsigned short* wpack = bpack + (size_t)NSEQ * NB;      // 32*128*32

  pack_wout<<<(KAB * CZ) / 256, 256, 0, stream>>>(Wout, wpack);
  opm_ln_proj<<<(NSEQ * NRES) / 8, 256, 0, stream>>>(m, gamma, beta, W1, b1,
                                                     W2, b2, aT, bpack);
  opm_gemm<<<dim3(NRES / 16, NRES), 256, 0, stream>>>(aT, bpack, wpack, bout,
                                                      out);
}
